// Prismion_51823075394176
// MI455X (gfx1250) — matrix operand order repaired and checked by reading, not compiled, not run on hardware
//
#include <hip/hip_runtime.h>
#include <math.h>

typedef __attribute__((ext_vector_type(16))) _Float16 v16h;
typedef __attribute__((ext_vector_type(8)))  float    v8f;
typedef __attribute__((ext_vector_type(4)))  float    f4;

static constexpr int BATCH = 1024;
static constexpr int INP   = 512;
static constexpr int MSGD  = 256;
static constexpr int SLOTD = 512;
static constexpr int RRD   = 256;
static constexpr int NW    = 9;      // 2K+1, K=4
static constexpr float GTAU = 2.0f;
static constexpr float BETA = 0.9f;

__device__ __forceinline__ float pymod(float a, float m) {
    float r = fmodf(a, m);
    if (r < 0.0f) r += m;
    return r;
}
__device__ __forceinline__ float wrap_delta(float a, float b, float rr) {
    return pymod(b - a + rr * 0.5f, rr) - rr * 0.5f;
}
__device__ __forceinline__ float circ_lerp(float a, float b, float w, float rr) {
    return pymod(a + w * wrap_delta(a, b, rr), rr);
}
__device__ __forceinline__ float sigm(float x) { return 1.0f / (1.0f + expf(-x)); }
__device__ __forceinline__ float softplusf(float x) { return (x > 20.0f) ? x : log1pf(expf(x)); }

// ---------------------------------------------------------------------------
// Kernel 0: per-batch ring gather: normalized ptr, softmax weights, centers,
// cur[b,d] = sum_k w[k]*ring[b,c[k],d]
// ---------------------------------------------------------------------------
__global__ void prep_kernel(const float* __restrict__ ring,
                            const float* __restrict__ ptr,
                            float* __restrict__ cur,
                            float* __restrict__ wts,
                            int*   __restrict__ centers,
                            float* __restrict__ ptrn) {
    const int b = blockIdx.x;
    const int tid = threadIdx.x;
    const float rr = (float)RRD;

    float praw = ptr[b];
    float pv;
    if (isnan(praw))       pv = 0.0f;
    else if (isinf(praw))  pv = (praw > 0.0f) ? rr - 1.0f : 0.0f;
    else                   pv = praw;
    pv = pymod(pv, rr);

    int base = (int)floorf(pv);
    base = min(max(base, 0), RRD - 1);

    float w[NW];
    int   c[NW];
    float mx = -1e30f;
#pragma unroll
    for (int k = 0; k < NW; k++) {
        int cc = base + (k - 4);
        cc = ((cc % RRD) + RRD) % RRD;
        c[k] = cc;
        float d = wrap_delta(pv, (float)cc, rr);
        float e = -(d * d) / GTAU;
        w[k] = e;
        mx = fmaxf(mx, e);
    }
    float s = 0.0f;
#pragma unroll
    for (int k = 0; k < NW; k++) { w[k] = expf(w[k] - mx); s += w[k]; }
    float inv = 1.0f / s;
#pragma unroll
    for (int k = 0; k < NW; k++) w[k] *= inv;

    if (tid < NW) { wts[b * NW + tid] = w[tid]; centers[b * NW + tid] = c[tid]; }
    if (tid == 0) ptrn[b] = pv;

    const float* rb = ring + (size_t)b * RRD * SLOTD;
    for (int d = tid; d < SLOTD; d += blockDim.x) {
        float acc = 0.0f;
#pragma unroll
        for (int k = 0; k < NW; k++) acc += w[k] * rb[(size_t)c[k] * SLOTD + d];
        cur[(size_t)b * SLOTD + d] = acc;
    }
}

// ---------------------------------------------------------------------------
// Kernel 1: h_pre = x @ Wi^T via WMMA f16 (f32 accumulate); fused epilogue:
// inp=tanh(+bi); cue=inp+sig(ctx)*cur; h_new=tanh(cue+h); h_out=mask?h_new:h
// One wave (32 threads) per 16x16 output tile.
// ---------------------------------------------------------------------------
__global__ void gemm1_kernel(const float* __restrict__ X,
                             const float* __restrict__ Wi,
                             const float* __restrict__ bi,
                             const float* __restrict__ cur,
                             const float* __restrict__ h,
                             const unsigned char* __restrict__ am,
                             const float* __restrict__ ctx_logit,
                             float* __restrict__ h_out) {
    const int ntiles_n = SLOTD / 16;                // 32
    const int m0 = (blockIdx.x / ntiles_n) * 16;
    const int n0 = (blockIdx.x % ntiles_n) * 16;
    const int lane = threadIdx.x;
    const int hi = lane >> 4;                        // 0 or 1
    const int lm = lane & 15;

    v8f c = {};
    const int arow = m0 + lm;
    const int brow = n0 + lm;

    for (int k0 = 0; k0 < INP; k0 += 32) {
        v16h a, bf;
        // A: lanes 0-15 hold K k0+0..7 & k0+16..23; lanes 16-31 hold K k0+8..15 & k0+24..31
        const float* ap = X + (size_t)arow * INP + (k0 + hi * 8);
        f4 a0 = *(const f4*)(ap + 0);
        f4 a1 = *(const f4*)(ap + 4);
        f4 a2 = *(const f4*)(ap + 16);
        f4 a3 = *(const f4*)(ap + 20);
#pragma unroll
        for (int j = 0; j < 4; j++) {
            a[j]      = (_Float16)a0[j];
            a[4 + j]  = (_Float16)a1[j];
            a[8 + j]  = (_Float16)a2[j];
            a[12 + j] = (_Float16)a3[j];
        }
        // B (= Wi^T): lane = N; lanes 0-15 hold K k0+0..7 & k0+16..23; lanes 16-31 hold K k0+8..15 & k0+24..31
        const float* bp = Wi + (size_t)brow * INP + (k0 + hi * 8);
        f4 b0 = *(const f4*)(bp + 0);
        f4 b1 = *(const f4*)(bp + 4);
        f4 b2 = *(const f4*)(bp + 16);
        f4 b3 = *(const f4*)(bp + 20);
#pragma unroll
        for (int j = 0; j < 4; j++) {
            bf[j]      = (_Float16)b0[j];
            bf[4 + j]  = (_Float16)b1[j];
            bf[8 + j]  = (_Float16)b2[j];
            bf[12 + j] = (_Float16)b3[j];
        }
        c = __builtin_amdgcn_wmma_f32_16x16x32_f16(false, a, false, bf,
                                                   (short)0, c, false, false);
    }

    const float sg = sigm(ctx_logit[0]);
#pragma unroll
    for (int r = 0; r < 8; r++) {
        const int m = m0 + r + hi * 8;
        const int n = n0 + lm;
        const float pre = c[r] + bi[n];
        const float inp = tanhf(pre);
        const float cue = inp + sg * cur[(size_t)m * SLOTD + n];
        const float hv  = h[(size_t)m * SLOTD + n];
        const float hn  = tanhf(cue + hv);
        h_out[(size_t)m * SLOTD + n] = am[m] ? hn : hv;
    }
}

// ---------------------------------------------------------------------------
// Kernel 2: per-batch scalar head: 5 dot products + circular pointer update.
// ---------------------------------------------------------------------------
__global__ void scalars_kernel(const float* __restrict__ upd,
                               const float* __restrict__ cur,
                               const float* __restrict__ Wj, const float* __restrict__ bj,
                               const float* __restrict__ Wn, const float* __restrict__ bn,
                               const float* __restrict__ Wd, const float* __restrict__ bd,
                               const float* __restrict__ Ww, const float* __restrict__ bw,
                               const float* __restrict__ Wg, const float* __restrict__ bg,
                               const float* __restrict__ theta_ptr,
                               const float* __restrict__ theta_gate,
                               const float* __restrict__ ptrn,
                               const unsigned char* __restrict__ am,
                               const float* __restrict__ gain_prev,
                               const float* __restrict__ gain_ema,
                               float* __restrict__ ptr_out,
                               float* __restrict__ gain_ws,
                               float* __restrict__ gain_prev_out,
                               float* __restrict__ gain_ema_out) {
    const int b = blockIdx.x;
    const int tid = threadIdx.x;
    __shared__ float red[5 * 256];

    float sj = 0, sn = 0, sd = 0, sw = 0, sg = 0;
    const float* u  = upd + (size_t)b * SLOTD;
    const float* cu = cur + (size_t)b * SLOTD;
    for (int d = tid; d < SLOTD; d += 256) {
        const float uv = u[d];
        sj += uv * Wj[d];
        sn += uv * Wn[d];
        sd += uv * Wd[d];
        sw += uv * Ww[d];
        sg += cu[d] * Wg[d] + uv * Wg[SLOTD + d];
    }
    red[0 * 256 + tid] = sj;
    red[1 * 256 + tid] = sn;
    red[2 * 256 + tid] = sd;
    red[3 * 256 + tid] = sw;
    red[4 * 256 + tid] = sg;
    __syncthreads();
    for (int s = 128; s > 0; s >>= 1) {
        if (tid < s) {
#pragma unroll
            for (int q = 0; q < 5; q++) red[q * 256 + tid] += red[q * 256 + tid + s];
        }
        __syncthreads();
    }

    if (tid == 0) {
        const float rr = (float)RRD;
        const float inertia = fminf(fmaxf(sigm(red[1 * 256] + bn[0]), 0.0f), 0.99f);
        const float dead    = softplusf(red[2 * 256] + bd[0]);
        const float walk    = fminf(fmaxf(sigm(red[3 * 256] + bw[0]), 0.0f), 1.0f);
        const float prev    = ptrn[b];
        int pbase = (int)floorf(prev);
        pbase = min(max(pbase, 0), RRD - 1);
        const float tp = sigm(theta_ptr[pbase]) * (rr - 1.0f);
        const float tg = theta_gate[pbase];
        const float p  = sigm(red[0 * 256] + bj[0] + tg);
        const float target   = rintf(tp);
        const float walk_ptr = pymod(prev + 1.0f, rr);
        float pn = circ_lerp(prev, walk_ptr, walk, rr);
        pn = circ_lerp(pn, target, p, rr);
        pn = circ_lerp(prev, pn, 1.0f - inertia, rr);
        const float dr   = wrap_delta(prev, pn, rr);
        const float move = sigm((fabsf(dr) - dead) * 4.0f);
        pn = pymod(prev + move * dr, rr);
        ptr_out[b] = am[b] ? pn : prev;

        const float g = sigm(red[4 * 256] + bg[0]);
        gain_ws[b] = g;
        gain_prev_out[b] = am[b] ? g : gain_prev[b];
        gain_ema_out[b]  = am[b] ? (BETA * gain_ema[b] + (1.0f - BETA) * g) : gain_ema[b];
    }
}

// ---------------------------------------------------------------------------
// Kernel 3: msg = tanh([cur|upd] @ Wm^T + bm) * gain, via WMMA; fused
// msg_prev / msg_ema epilogues. One wave per 16x16 tile.
// ---------------------------------------------------------------------------
__global__ void gemm2_kernel(const float* __restrict__ cur,
                             const float* __restrict__ upd,
                             const float* __restrict__ Wm,
                             const float* __restrict__ bm,
                             const float* __restrict__ gain,
                             const unsigned char* __restrict__ am,
                             const float* __restrict__ msg_prev,
                             const float* __restrict__ msg_ema,
                             float* __restrict__ msg_out,
                             float* __restrict__ msg_prev_out,
                             float* __restrict__ msg_ema_out) {
    const int ntiles_n = MSGD / 16;                  // 16
    const int m0 = (blockIdx.x / ntiles_n) * 16;
    const int n0 = (blockIdx.x % ntiles_n) * 16;
    const int lane = threadIdx.x;
    const int hi = lane >> 4;
    const int lm = lane & 15;

    v8f c = {};
    const int arow = m0 + lm;
    const int brow = n0 + lm;

    for (int k0 = 0; k0 < 2 * SLOTD; k0 += 32) {
        // rw = [cur | upd]; a K-tile never straddles the 512 boundary
        const float* abase = (k0 < SLOTD)
            ? (cur + (size_t)arow * SLOTD + k0)
            : (upd + (size_t)arow * SLOTD + (k0 - SLOTD));
        v16h a, bf;
        const float* ap = abase + hi * 8;
        f4 a0 = *(const f4*)(ap + 0);
        f4 a1 = *(const f4*)(ap + 4);
        f4 a2 = *(const f4*)(ap + 16);
        f4 a3 = *(const f4*)(ap + 20);
#pragma unroll
        for (int j = 0; j < 4; j++) {
            a[j]      = (_Float16)a0[j];
            a[4 + j]  = (_Float16)a1[j];
            a[8 + j]  = (_Float16)a2[j];
            a[12 + j] = (_Float16)a3[j];
        }
        const float* bp = Wm + (size_t)brow * (2 * SLOTD) + k0 + hi * 8;
        f4 b0 = *(const f4*)(bp + 0);
        f4 b1 = *(const f4*)(bp + 4);
        f4 b2 = *(const f4*)(bp + 16);
        f4 b3 = *(const f4*)(bp + 20);
#pragma unroll
        for (int j = 0; j < 4; j++) {
            bf[j]      = (_Float16)b0[j];
            bf[4 + j]  = (_Float16)b1[j];
            bf[8 + j]  = (_Float16)b2[j];
            bf[12 + j] = (_Float16)b3[j];
        }
        c = __builtin_amdgcn_wmma_f32_16x16x32_f16(false, a, false, bf,
                                                   (short)0, c, false, false);
    }

#pragma unroll
    for (int r = 0; r < 8; r++) {
        const int m = m0 + r + hi * 8;
        const int n = n0 + lm;
        const size_t idx = (size_t)m * MSGD + n;
        const float msg = tanhf(c[r] + bm[n]);
        const float mo  = msg * gain[m];
        msg_out[idx] = mo;
        const float mp = msg_prev[idx];
        const float me = msg_ema[idx];
        const bool  a_ = am[m] != 0;
        msg_prev_out[idx] = a_ ? mo : mp;
        msg_ema_out[idx]  = a_ ? (BETA * me + (1.0f - BETA) * mo) : me;
    }
}

// ---------------------------------------------------------------------------
// Kernel 4: streaming non-temporal b128 copy of ring -> ring_out (512 MB).
// ---------------------------------------------------------------------------
__global__ void copyring_kernel(const f4* __restrict__ src,
                                f4* __restrict__ dst, size_t n4) {
    size_t i = (size_t)blockIdx.x * blockDim.x + threadIdx.x;
    const size_t stride = (size_t)gridDim.x * blockDim.x;
    for (; i < n4; i += stride) {
        f4 v = __builtin_nontemporal_load(src + i);
        __builtin_nontemporal_store(v, dst + i);
    }
}

// ---------------------------------------------------------------------------
// Kernel 5: scatter-add contributions into ring_out at the 9 centers per row.
// ---------------------------------------------------------------------------
__global__ void scatter_kernel(float* __restrict__ ring_out,
                               const float* __restrict__ upd,
                               const float* __restrict__ wts,
                               const int* __restrict__ centers,
                               const unsigned char* __restrict__ am) {
    const int b = blockIdx.x;
    if (!am[b]) return;                 // amf == 0 -> no contribution
    float w[NW];
    int   c[NW];
#pragma unroll
    for (int k = 0; k < NW; k++) { w[k] = wts[b * NW + k]; c[k] = centers[b * NW + k]; }
    float* rb = ring_out + (size_t)b * RRD * SLOTD;
    const float* u = upd + (size_t)b * SLOTD;
    for (int d = threadIdx.x; d < SLOTD; d += blockDim.x) {
        const float uv = u[d];
#pragma unroll
        for (int k = 0; k < NW; k++) rb[(size_t)c[k] * SLOTD + d] += w[k] * uv;
    }
}

// ---------------------------------------------------------------------------
extern "C" void kernel_launch(void* const* d_in, const int* in_sizes, int n_in,
                              void* d_out, int out_size, void* d_ws, size_t ws_size,
                              hipStream_t stream) {
    const float* x          = (const float*)d_in[0];
    const float* ring       = (const float*)d_in[1];
    const float* ptr        = (const float*)d_in[2];
    const float* h          = (const float*)d_in[3];
    const float* msg_prev   = (const float*)d_in[4];
    const float* msg_ema    = (const float*)d_in[5];
    const float* gain_prev  = (const float*)d_in[6];
    const float* gain_ema   = (const float*)d_in[7];
    const float* Wi         = (const float*)d_in[8];
    const float* bi         = (const float*)d_in[9];
    const float* Wj         = (const float*)d_in[10];
    const float* bj         = (const float*)d_in[11];
    const float* Wn         = (const float*)d_in[12];
    const float* bn         = (const float*)d_in[13];
    const float* Wd         = (const float*)d_in[14];
    const float* bd         = (const float*)d_in[15];
    const float* Ww         = (const float*)d_in[16];
    const float* bw         = (const float*)d_in[17];
    const float* theta_ptr  = (const float*)d_in[18];
    const float* theta_gate = (const float*)d_in[19];
    const float* ctx        = (const float*)d_in[20];
    const float* Wm         = (const float*)d_in[21];
    const float* bm         = (const float*)d_in[22];
    const float* Wg         = (const float*)d_in[23];
    const float* bg         = (const float*)d_in[24];
    const unsigned char* am = (const unsigned char*)d_in[25];

    // Output layout (flat, return order)
    float* out           = (float*)d_out;
    float* msg_out       = out;                                     // B*MSG
    float* ring_out      = msg_out + (size_t)BATCH * MSGD;          // B*RR*SLOT
    float* ptr_out       = ring_out + (size_t)BATCH * RRD * SLOTD;  // B
    float* h_out         = ptr_out + BATCH;                         // B*SLOT
    float* msg_prev_out  = h_out + (size_t)BATCH * SLOTD;           // B*MSG
    float* msg_ema_out   = msg_prev_out + (size_t)BATCH * MSGD;     // B*MSG
    float* gain_prev_out = msg_ema_out + (size_t)BATCH * MSGD;      // B
    float* gain_ema_out  = gain_prev_out + BATCH;                   // B

    // Workspace layout
    float* ws      = (float*)d_ws;
    float* cur     = ws;                                 // B*SLOT
    float* wts     = cur + (size_t)BATCH * SLOTD;        // B*NW
    int*   centers = (int*)(wts + BATCH * NW);           // B*NW
    float* ptrn    = (float*)(centers + BATCH * NW);     // B
    float* gainw   = ptrn + BATCH;                       // B

    prep_kernel<<<BATCH, 256, 0, stream>>>(ring, ptr, cur, wts, centers, ptrn);

    gemm1_kernel<<<(BATCH / 16) * (SLOTD / 16), 32, 0, stream>>>(
        x, Wi, bi, cur, h, am, ctx, h_out);

    scalars_kernel<<<BATCH, 256, 0, stream>>>(
        h_out, cur, Wj, bj, Wn, bn, Wd, bd, Ww, bw, Wg, bg,
        theta_ptr, theta_gate, ptrn, am, gain_prev, gain_ema,
        ptr_out, gainw, gain_prev_out, gain_ema_out);

    gemm2_kernel<<<(BATCH / 16) * (MSGD / 16), 32, 0, stream>>>(
        cur, h_out, Wm, bm, gainw, am, msg_prev, msg_ema,
        msg_out, msg_prev_out, msg_ema_out);

    const size_t n4 = (size_t)BATCH * RRD * SLOTD / 4;
    copyring_kernel<<<4096, 256, 0, stream>>>((const f4*)ring, (f4*)ring_out, n4);

    scatter_kernel<<<BATCH, 256, 0, stream>>>(ring_out, h_out, wts, centers, am);
}
